// AE_29171417875247
// MI455X (gfx1250) — matrix operand order repaired and checked by reading, not compiled, not run on hardware
//
#include <hip/hip_runtime.h>
#include <hip/hip_bf16.h>

typedef __attribute__((ext_vector_type(16))) __bf16       v16bf;
typedef __attribute__((ext_vector_type(2)))  __bf16       v2bf;
typedef __attribute__((ext_vector_type(8)))  float        v8f;
typedef __attribute__((ext_vector_type(4)))  unsigned int u32x4;
typedef __attribute__((ext_vector_type(4)))  int          i32x4;
typedef __attribute__((ext_vector_type(8)))  int          i32x8;

#if defined(__has_builtin)
#if __has_builtin(__builtin_amdgcn_tensor_load_to_lds)
#define TDM_OK 1
#endif
#if __has_builtin(__builtin_amdgcn_cvt_pk_bf16_f32)
#define PKCVT_OK 1
#endif
#endif
#ifndef TDM_OK
#define TDM_OK 0
#endif
#ifndef PKCVT_OK
#define PKCVT_OK 0
#endif

struct Q2 { uint4 q0, q1; };
union Frag { Q2 q; v16bf v; };

__device__ __forceinline__ unsigned pack_bf16x2(float lo, float hi) {
#if PKCVT_OK
    v2bf r = __builtin_amdgcn_cvt_pk_bf16_f32(lo, hi);
    return __builtin_bit_cast(unsigned, r);
#else
    unsigned ul = __float_as_uint(lo), uh = __float_as_uint(hi);
    ul += 0x7FFFu + ((ul >> 16) & 1u);
    uh += 0x7FFFu + ((uh >> 16) & 1u);
    return (ul >> 16) | (uh & 0xFFFF0000u);
#endif
}
__device__ __forceinline__ unsigned short bf16_rne(float f) {
#if PKCVT_OK
    return (unsigned short)(pack_bf16x2(f, f) & 0xFFFFu);
#else
    unsigned u = __float_as_uint(f);
    u += 0x7FFFu + ((u >> 16) & 1u);
    return (unsigned short)(u >> 16);
#endif
}

#define BM 128
#define BN 128
#define BK 32
#define LDA_B 80                    // LDS bytes per tile row (64B data + 16B pad)
#define A_BYTES (BM * LDA_B)        // 10240
#define B_BYTES (BN * LDA_B)        // 10240 (B tile identical shape: 128 rows x 64B)
#define BUF_BYTES (A_BYTES + B_BYTES)

#if TDM_OK
// One TDM descriptor loads a full 128-row x 32-bf16-col tile (row stride = K
// elements) from global into LDS with an 80B LDS row pitch:
//   pad_enable=1, pad_interval=3 (pad after 16 DWORDs=64B), pad_amount=3 (4 DWORDs=16B)
__device__ __forceinline__ void tdm_tile_bf16(const void* gbase, unsigned lds_off,
                                              int row0, int col0, int K, int ROWS) {
    unsigned long long ga = (unsigned long long)gbase +
                            ((unsigned long long)row0 * (unsigned)K + (unsigned)col0) * 2ull;
    u32x4 g0;
    g0.x = 1u;                                        // count=1, user descriptor
    g0.y = lds_off;                                   // D#.lds_addr
    g0.z = (unsigned)(ga & 0xFFFFFFFFull);            // global_addr[31:0]
    g0.w = (unsigned)(ga >> 32) | (2u << 30);         // global_addr[56:32] | type=2
    i32x8 g1;
    g1[0] = (int)((1u << 16) | (1u << 20) | (3u << 22) | (3u << 25));
            // data_size=2B | pad_enable | pad_interval=3 | pad_amount=3
    g1[1] = (int)(((unsigned)K & 0xFFFFu) << 16);                          // tensor_dim0 lo
    g1[2] = (int)((((unsigned)K >> 16) & 0xFFFFu) |
                  (((unsigned)ROWS & 0xFFFFu) << 16));                     // dim0 hi | dim1 lo
    g1[3] = (int)((((unsigned)ROWS >> 16) & 0xFFFFu) | (32u << 16));       // dim1 hi | tile_dim0=32
    g1[4] = 128;                                                           // tile_dim1=128
    g1[5] = (int)(unsigned)K;                                              // tensor_dim0_stride
    g1[6] = 0;
    g1[7] = 0;
    i32x4 z4 = {0, 0, 0, 0};
#if __clang_major__ >= 23
    i32x8 z8 = {0, 0, 0, 0, 0, 0, 0, 0};
    __builtin_amdgcn_tensor_load_to_lds(g0, g1, z4, z4, z8, 0);
#else
    __builtin_amdgcn_tensor_load_to_lds(g0, g1, z4, z4, 0);
#endif
}
#endif

// C = act(A[M,K] * Bt^T + bias): A bf16 (or f32, converted in-flight) row-major,
// Bt = bf16 transposed weights [N][K]. WMMA 16x16x32 bf16, f32 accumulate.
// 128x128x32 tiles, 8 wave32 per block, each wave owns 64x32 (4x2 WMMA tiles).
// Double-buffered LDS, ONE barrier per K-step. A/B tiles DMA'd by the TDM
// (wave0/wave1) when available.
template<bool ABF16, bool SIG, bool OBF16>
__global__ __launch_bounds__(256)
void gemm_bias_act(const void* __restrict__ Aptr,
                   const unsigned short* __restrict__ Bt,
                   const float* __restrict__ bias,
                   void* __restrict__ Out,
                   int M, int N, int K)
{
    __shared__ __align__(16) char smem[2 * BUF_BYTES];

    const int tid  = threadIdx.x;
    const int m0   = blockIdx.y * BM;
    const int n0   = blockIdx.x * BN;
    const int lane = tid & 31, l16 = lane & 15, hi = lane >> 4;
    const int wave = tid >> 5, wm = wave & 1, wn = wave >> 1;
    const int arow = tid >> 1, ahalf = tid & 1;   // manual staging: 2 threads/row, 32B each

    float4 af[4];          // f32 A staging
    uint4  aqA[2], aqB[2]; // bf16 copy staging (fallback)

    v8f acc[4][2];
#pragma unroll
    for (int i = 0; i < 4; ++i)
#pragma unroll
        for (int j = 0; j < 2; ++j)
#pragma unroll
            for (int e = 0; e < 8; ++e) acc[i][j][e] = 0.0f;

    const unsigned lds0 = (unsigned)(unsigned long long)(uintptr_t)&smem[0];

    // Issue phase: DMA/global->reg for tile kt into buffer `buf` (runs before compute).
    auto stage_issue = [&](int kt, int buf) {
#if TDM_OK
        if (ABF16) {
            if (wave == 0)
                tdm_tile_bf16(Aptr, lds0 + buf * BUF_BYTES, m0, kt * BK, K, M);
        } else {
            const float4* ap = (const float4*)((const float*)Aptr +
                               (size_t)(m0 + arow) * K + kt * BK + ahalf * 16);
#pragma unroll
            for (int j = 0; j < 4; ++j) af[j] = ap[j];
            __builtin_prefetch((const float*)Aptr + (size_t)(m0 + arow) * K + (kt + 1) * BK, 0, 1);
        }
        if (wave == 1)
            tdm_tile_bf16(Bt, lds0 + buf * BUF_BYTES + A_BYTES, n0, kt * BK, K, N);
#else
        if (ABF16) {
            const uint4* ap = (const uint4*)((const unsigned short*)Aptr +
                              (size_t)(m0 + arow) * K + kt * BK + ahalf * 16);
            aqA[0] = ap[0]; aqA[1] = ap[1];
        } else {
            const float4* ap = (const float4*)((const float*)Aptr +
                               (size_t)(m0 + arow) * K + kt * BK + ahalf * 16);
#pragma unroll
            for (int j = 0; j < 4; ++j) af[j] = ap[j];
        }
        const uint4* bp = (const uint4*)(Bt + (size_t)(n0 + arow) * K + kt * BK + ahalf * 16);
        aqB[0] = bp[0]; aqB[1] = bp[1];
#endif
        (void)buf;
    };

    // Commit phase: reg->LDS stores / TDM completion wait (runs after compute).
    auto stage_commit = [&](int buf) {
        char* a = smem + buf * BUF_BYTES;
        char* b = a + A_BYTES;
        if (!ABF16) {
            uint4 w0, w1;
            w0.x = pack_bf16x2(af[0].x, af[0].y); w0.y = pack_bf16x2(af[0].z, af[0].w);
            w0.z = pack_bf16x2(af[1].x, af[1].y); w0.w = pack_bf16x2(af[1].z, af[1].w);
            w1.x = pack_bf16x2(af[2].x, af[2].y); w1.y = pack_bf16x2(af[2].z, af[2].w);
            w1.z = pack_bf16x2(af[3].x, af[3].y); w1.w = pack_bf16x2(af[3].z, af[3].w);
            uint4* ad = (uint4*)(a + arow * LDA_B + ahalf * 32);
            ad[0] = w0; ad[1] = w1;
        }
#if TDM_OK
        if ((ABF16 && wave == 0) || wave == 1)
            __builtin_amdgcn_s_wait_tensorcnt(0);
#else
        if (ABF16) {
            uint4* ad = (uint4*)(a + arow * LDA_B + ahalf * 32);
            ad[0] = aqA[0]; ad[1] = aqA[1];
        }
        uint4* bd = (uint4*)(b + arow * LDA_B + ahalf * 32);
        bd[0] = aqB[0]; bd[1] = aqB[1];
#endif
        (void)b;
    };

    auto compute = [&](int buf) {
        const char* a = smem + buf * BUF_BYTES;
        const char* b = a + A_BYTES;
        Frag afr[4], bfr[2];
        // A frag (ISA 16x32 layout): lanes 0-15 K{0..7,16..23}, lanes 16-31 K{8..15,24..31}
#pragma unroll
        for (int mt = 0; mt < 4; ++mt) {
            int r = wm * 64 + mt * 16 + l16;
            const uint4* p = (const uint4*)(a + r * LDA_B + hi * 16);
            afr[mt].q.q0 = p[0];
            afr[mt].q.q1 = p[2];   // +32 bytes
        }
        // B frag (ISA 32x16 layout): lanes 0-15 K{0..7,16..23}, lanes 16-31 K{8..15,24..31}
#pragma unroll
        for (int nt = 0; nt < 2; ++nt) {
            int n = wn * 32 + nt * 16 + l16;
            const uint4* p = (const uint4*)(b + n * LDA_B + hi * 16);
            bfr[nt].q.q0 = p[0];
            bfr[nt].q.q1 = p[2];   // +32 bytes
        }
#pragma unroll
        for (int mt = 0; mt < 4; ++mt)
#pragma unroll
            for (int nt = 0; nt < 2; ++nt)
                acc[mt][nt] = __builtin_amdgcn_wmma_f32_16x16x32_bf16(
                    false, afr[mt].v, false, bfr[nt].v, (short)0, acc[mt][nt], false, false);
    };

    const int KT = K / BK;
    stage_issue(0, 0);
    stage_commit(0);
    __syncthreads();
    for (int kt = 0; kt < KT; ++kt) {
        const int cur = kt & 1, nxt = cur ^ 1;
        if (kt + 1 < KT) stage_issue(kt + 1, nxt);   // writes into nxt: last read at kt-1, ordered by that barrier
        compute(cur);
        if (kt + 1 < KT) stage_commit(nxt);
        __syncthreads();                              // publishes nxt, guards cur against kt+1's stores
    }

    int   coln[2];
    float bs[2];
#pragma unroll
    for (int nt = 0; nt < 2; ++nt) {
        coln[nt] = n0 + wn * 32 + nt * 16 + l16;
        bs[nt]   = bias[coln[nt]];
    }
#pragma unroll
    for (int mt = 0; mt < 4; ++mt) {
        int rbase = m0 + wm * 64 + mt * 16 + hi * 8;
#pragma unroll
        for (int nt = 0; nt < 2; ++nt) {
#pragma unroll
            for (int i = 0; i < 8; ++i) {
                float v = acc[mt][nt][i] + bs[nt];
                if (SIG) v = 1.0f / (1.0f + __expf(-v));
                size_t off = (size_t)(rbase + i) * N + coln[nt];
                if (OBF16) ((unsigned short*)Out)[off] = bf16_rne(v);
                else       ((float*)Out)[off] = v;
            }
        }
    }
}

// W[K][N] f32 -> Wt[N][K] bf16 (transpose + convert), LDS-tiled 64x64.
__global__ __launch_bounds__(256)
void transpose_pack_bf16(const float* __restrict__ W, unsigned short* __restrict__ Wt,
                         int K, int N)
{
    __shared__ float tile[64][65];
    const int k0 = blockIdx.y * 64, n0 = blockIdx.x * 64;
    const int t = threadIdx.x, rr = t >> 2, part = t & 3;

    const float4* src = (const float4*)&W[(size_t)(k0 + rr) * N + n0 + part * 16];
#pragma unroll
    for (int j = 0; j < 4; ++j) {
        float4 f = src[j];
        tile[rr][part * 16 + 4 * j + 0] = f.x;
        tile[rr][part * 16 + 4 * j + 1] = f.y;
        tile[rr][part * 16 + 4 * j + 2] = f.z;
        tile[rr][part * 16 + 4 * j + 3] = f.w;
    }
    __syncthreads();

    unsigned w[8];
#pragma unroll
    for (int j = 0; j < 8; ++j)
        w[j] = pack_bf16x2(tile[part * 16 + 2 * j][rr], tile[part * 16 + 2 * j + 1][rr]);
    uint4* dst = (uint4*)&Wt[(size_t)(n0 + rr) * K + k0 + part * 16];
    uint4 q0, q1;
    q0.x = w[0]; q0.y = w[1]; q0.z = w[2]; q0.w = w[3];
    q1.x = w[4]; q1.y = w[5]; q1.z = w[6]; q1.w = w[7];
    dst[0] = q0; dst[1] = q1;
}

// Per-row top-k on sigmoid outputs in (0,1): IEEE bits are order-preserving ->
// 30-step bitwise binary search for the k-th largest, then masked bf16 emit.
__global__ __launch_bounds__(256)
void topk_mask(const float* __restrict__ h, unsigned short* __restrict__ out, int N, int k)
{
    __shared__ float vals[1024];
    __shared__ int   cnt[256];
    const int    tid    = threadIdx.x;
    const size_t rowoff = (size_t)blockIdx.x * N;

    for (int j = tid; j < N; j += 256) vals[j] = h[rowoff + j];
    __syncthreads();

    unsigned tau = 0u;
    for (int b = 29; b >= 0; --b) {
        unsigned cand = tau | (1u << b);
        float    t    = __uint_as_float(cand);
        int c = 0;
        for (int j = tid; j < N; j += 256) c += (vals[j] >= t) ? 1 : 0;
        cnt[tid] = c;
        __syncthreads();
        for (int s = 128; s > 0; s >>= 1) {
            if (tid < s) cnt[tid] += cnt[tid + s];
            __syncthreads();
        }
        int total = cnt[0];
        __syncthreads();
        if (total >= k) tau = cand;
    }
    float t = __uint_as_float(tau);
    for (int j = tid; j < N; j += 256) {
        float v = vals[j];
        out[rowoff + j] = (v >= t) ? bf16_rne(v) : (unsigned short)0;
    }
}

extern "C" void kernel_launch(void* const* d_in, const int* in_sizes, int n_in,
                              void* d_out, int out_size, void* d_ws, size_t ws_size,
                              hipStream_t stream) {
    (void)in_sizes; (void)n_in; (void)out_size; (void)ws_size;
    const float* x   = (const float*)d_in[0];
    const float* eW0 = (const float*)d_in[1];
    const float* eb0 = (const float*)d_in[2];
    const float* eW1 = (const float*)d_in[3];
    const float* eb1 = (const float*)d_in[4];
    const float* dW1 = (const float*)d_in[5];
    const float* db1 = (const float*)d_in[6];
    const float* dW0 = (const float*)d_in[7];
    const float* db0 = (const float*)d_in[8];

    const int B = 8192, D = 4096, H1 = 2048, H2 = 1024;
    const int K_TOP = (int)(H2 * 0.05);  // 51

    char* ws = (char*)d_ws;
    size_t off = 0;
    unsigned short* W0t  = (unsigned short*)(ws + off); off += (size_t)D  * H1 * 2;
    unsigned short* W1t  = (unsigned short*)(ws + off); off += (size_t)H1 * H2 * 2;
    unsigned short* dW1t = (unsigned short*)(ws + off); off += (size_t)H2 * H1 * 2;
    unsigned short* dW0t = (unsigned short*)(ws + off); off += (size_t)H1 * D  * 2;
    unsigned short* h1b  = (unsigned short*)(ws + off); off += (size_t)B  * H1 * 2;
    float*          h2f  = (float*)         (ws + off); off += (size_t)B  * H2 * 4;
    unsigned short* h2mb = (unsigned short*)(ws + off); off += (size_t)B  * H2 * 2;
    unsigned short* dmb  = (unsigned short*)(ws + off); off += (size_t)B  * H1 * 2;

    dim3 blk(256);
    // Weight transpose+convert pre-pass (bf16 weights then live in the 192MB L2).
    transpose_pack_bf16<<<dim3(H1 / 64, D  / 64), blk, 0, stream>>>(eW0, W0t,  D,  H1);
    transpose_pack_bf16<<<dim3(H2 / 64, H1 / 64), blk, 0, stream>>>(eW1, W1t,  H1, H2);
    transpose_pack_bf16<<<dim3(H1 / 64, H2 / 64), blk, 0, stream>>>(dW1, dW1t, H2, H1);
    transpose_pack_bf16<<<dim3(D  / 64, H1 / 64), blk, 0, stream>>>(dW0, dW0t, H1, D);

    // h1 = sigmoid(x @ enc_W0 + b0): f32 A converted in-flight, bf16 out
    gemm_bias_act<false, true,  true ><<<dim3(H1 / BN, B / BM), blk, 0, stream>>>(
        x, W0t, eb0, h1b, B, H1, D);
    // h2 = sigmoid(h1 @ enc_W1 + b1): TDM A+B, f32 out for top-k threshold search
    gemm_bias_act<true,  true,  false><<<dim3(H2 / BN, B / BM), blk, 0, stream>>>(
        h1b, W1t, eb1, h2f, B, H2, H1);
    topk_mask<<<dim3(B), blk, 0, stream>>>(h2f, h2mb, H2, K_TOP);
    // d = sigmoid(h2m @ dec_W1 + b1)
    gemm_bias_act<true,  true,  true ><<<dim3(H1 / BN, B / BM), blk, 0, stream>>>(
        h2mb, dW1t, db1, dmb, B, H1, H2);
    // out = d @ dec_W0 + b0 (linear)
    gemm_bias_act<true,  false, false><<<dim3(D / BN, B / BM), blk, 0, stream>>>(
        dmb, dW0t, db0, (float*)d_out, B, D, H1);
}
